// GNNMorseModel_78563541778966
// MI455X (gfx1250) — matrix operand order repaired and checked by reading, not compiled, not run on hardware
//
#include <hip/hip_runtime.h>
#include <math.h>

typedef __attribute__((ext_vector_type(16))) _Float16 v16h;
typedef __attribute__((ext_vector_type(8)))  float    v8f;
typedef __attribute__((ext_vector_type(8)))  int      v8i;

#define D16 16

__device__ __forceinline__ float fastrcp(float x) { return __builtin_amdgcn_rcpf(x); }
__device__ __forceinline__ float siluf(float x) {
    return x * fastrcp(1.f + __expf(-x));      // v_rcp_f32 instead of IEEE div chain
}
__device__ __forceinline__ float softplusf(float x) {
    return (x > 20.f) ? x : __logf(1.f + __expf(x));
}

__device__ __forceinline__ v8i  h2i(v16h x) { union { v16h h; v8i i; } u; u.h = x; return u.i; }
__device__ __forceinline__ v16h i2h(v8i  x) { union { v16h h; v8i i; } u; u.i = x; return u.h; }
__device__ __forceinline__ int  pkh2(float a, float b) {
    union { _Float16 h[2]; int i; } u;
    u.h[0] = (_Float16)a; u.h[1] = (_Float16)b; return u.i;
}

// D = A*B + C  (16x16x32 f16 -> f32 accum)
__device__ __forceinline__ v8f wmma16(v16h a, v16h b, v8f c) {
    return __builtin_amdgcn_wmma_f32_16x16x32_f16(
        false, a, false, b, (short)0, c, false, false);
}

// Weight matrix W[nrows][16] into WMMA A layout (M=rows, K=cols, K padded 16->32).
// lane<16: row=lane, halfs 0..15 = W[row][0..15]; lane>=16: all halfs (K=16..31) = 0
// so B lanes 16..31 are don't-care.
__device__ __forceinline__ v16h wload_A16(const float* __restrict__ w, int lane, int nrows) {
    int row = lane & 15;
    int cb  = (lane >> 4) << 3;
    v16h a;
#pragma unroll
    for (int k = 0; k < 16; ++k)
        a[k] = (cb == 0 && row < nrows) ? (_Float16)w[row * 16 + k] : (_Float16)0.f;
    return a;
}

// B layout: lane n (0..15) holds x[n][k] at half k (K=0..15). Lanes 16..31 don't-care.
__device__ __forceinline__ v16h pack_B16(const float* x) {
    v16h b;
#pragma unroll
    for (int k = 0; k < 16; ++k) b[k] = (_Float16)x[k];
    return b;
}

// Tile-1 B: each lane fetches lane^16's packed B dwords (8 bpermutes, not 16).
__device__ __forceinline__ v16h shfl_B(v16h b, int lane) {
    v8i s = h2i(b), r;
#pragma unroll
    for (int j = 0; j < 8; ++j) r[j] = __shfl(s[j], lane ^ 16, 32);
    return i2h(r);
}

// Inter-layer B build: own act dims (0..7 for lane<16) in halfs 0..7, partner lane's
// (dims 8..15) in halfs 8..15 — 4 packed-dword bpermutes.
__device__ __forceinline__ v16h build_B2(const float* act, int lane) {
    int own[4], up[4];
#pragma unroll
    for (int j = 0; j < 4; ++j) own[j] = pkh2(act[2 * j], act[2 * j + 1]);
#pragma unroll
    for (int j = 0; j < 4; ++j) up[j] = __shfl(own[j], lane | 16, 32);
    v8i bi;
#pragma unroll
    for (int j = 0; j < 4; ++j) { bi[j] = own[j]; bi[j + 4] = up[j]; }
    return i2h(bi);
}

// ---------------- Kernel 1: embed gather + init accumulator + zero loss ----------------
__global__ void k_init(const int* __restrict__ elem, const float* __restrict__ embed,
                       float* __restrict__ h0, float* __restrict__ hacc,
                       float* __restrict__ loss_slot, int N) {
    int n = blockIdx.x * blockDim.x + threadIdx.x;
    if (n == 0) *loss_slot = 0.f;
    if (n >= N) return;
    int el = elem[n];
    const float4* er = (const float4*)(embed + el * D16);
    float4* a = (float4*)(h0 + (size_t)n * D16);
    float4* b = (float4*)(hacc + (size_t)n * D16);
#pragma unroll
    for (int q = 0; q < 4; ++q) { float4 v = er[q]; a[q] = v; b[q] = v; }
}

// ---------------- Kernel 2: RBF -> filter MLP (2x WMMA) -> messages -> scatter-add ------
__global__ void k_filter(const int* __restrict__ eidx, const float* __restrict__ dist,
                         const float* __restrict__ fw1, const float* __restrict__ fb1,
                         const float* __restrict__ fw2, const float* __restrict__ fb2,
                         const float* __restrict__ h0, float* __restrict__ hacc, int E) {
    const int lane = threadIdx.x & 31;
    const int wave = blockIdx.x * (blockDim.x >> 5) + (threadIdx.x >> 5);
    const int base = wave * 32;                 // 32 edges per wave (E % 32 == 0)
    if (base >= E) return;
    const int hi = lane >> 4;                   // 0: dims 0..7, 1: dims 8..15 in C

    v16h A1 = wload_A16(fw1, lane, 16);
    v16h A2 = wload_A16(fw2, lane, 16);

    // RBF for edge base+lane (one edge per lane)
    const float RMIN = 0.94486299430f;           // 0.5 * A2B
    const float STEP = 0.44093606400667f;        // (RMAX-RMIN)/15
    const float INVW = 1.0f / 0.41337756000625f; // 1 / ((RMAX-RMIN)/16)
    float d = dist[base + lane];
    float rbf[16];
#pragma unroll
    for (int k = 0; k < 16; ++k) {
        float t = (d - (RMIN + (float)k * STEP)) * INVW;
        rbf[k] = __expf(-0.5f * t * t);
    }
    v16h B0 = pack_B16(rbf);
    v16h B1 = shfl_B(B0, lane);                 // tile 1: edges base+16..base+31

    float b1v[8], b2v[8];
#pragma unroll
    for (int j = 0; j < 8; ++j) { b1v[j] = fb1[hi * 8 + j]; b2v[j] = fb2[hi * 8 + j]; }

#pragma unroll
    for (int t = 0; t < 2; ++t) {
        v8f cz = {};
        v8f c1 = wmma16(A1, t ? B1 : B0, cz);   // out1^T[dim][edge]
        float act[8];
#pragma unroll
        for (int j = 0; j < 8; ++j) act[j] = siluf(c1[j] + b1v[j]);
        v16h B2 = build_B2(act, lane);          // lane e: act dims 0..15 of edge e
        v8f c2 = wmma16(A2, B2, cz);            // W^T[dim][edge]

        int col  = lane & 15;
        int e2   = base + t * 16 + col;
        int srcn = eidx[e2];
        int tgtn = eidx[E + e2];
        const float4* hr = (const float4*)(h0 + (size_t)tgtn * D16 + hi * 8);
        float4 ha = hr[0], hb = hr[1];
        float hv[8] = {ha.x, ha.y, ha.z, ha.w, hb.x, hb.y, hb.z, hb.w};
        float* ap = hacc + (size_t)srcn * D16 + hi * 8;
#pragma unroll
        for (int j = 0; j < 8; ++j)
            atomicAdd(ap + j, hv[j] * (c2[j] + b2v[j]));
    }
}

// ---------------- Kernel 3: per-node VQ argmin + commitment loss ------------------------
__global__ void k_vq(const int* __restrict__ elem, const float* __restrict__ codebook,
                     const float* __restrict__ hacc, float* __restrict__ hq,
                     float* __restrict__ loss_slot, int N) {
    int n = blockIdx.x * blockDim.x + threadIdx.x;
    bool ok = n < N;
    float local = 0.f;
    if (ok) {
        float hv[16];
        const float4* hr = (const float4*)(hacc + (size_t)n * D16);
#pragma unroll
        for (int q = 0; q < 4; ++q) {
            float4 v = hr[q];
            hv[q * 4 + 0] = v.x; hv[q * 4 + 1] = v.y; hv[q * 4 + 2] = v.z; hv[q * 4 + 3] = v.w;
        }
        int el = elem[n];
        float hh = 0.f;
#pragma unroll
        for (int k = 0; k < 16; ++k) hh += hv[k] * hv[k];
        int best = el * 4; float bd = 3.4e38f;
#pragma unroll
        for (int c = 0; c < 4; ++c) {
            int code = el * 4 + c;
            const float* cb = codebook + code * 16;
            float dot = 0.f, cc = 0.f;
#pragma unroll
            for (int k = 0; k < 16; ++k) { float cv = cb[k]; dot += hv[k] * cv; cc += cv * cv; }
            float d2 = hh - 2.f * dot + cc;
            if (d2 < bd) { bd = d2; best = code; }   // first-min tie-break like argmin
        }
        const float* cb = codebook + best * 16;
        float* oq = hq + (size_t)n * D16;
#pragma unroll
        for (int k = 0; k < 16; ++k) {
            float cv = cb[k];
            oq[k] = cv;                               // h_q = z_q (straight-through fwd)
            float df = hv[k] - cv;
            local += df * df;
        }
    }
#pragma unroll
    for (int off = 16; off; off >>= 1) local += __shfl_down(local, off, 32);
    if ((threadIdx.x & 31) == 0)
        atomicAdd(loss_slot, local * (0.25f / ((float)N * 16.f)));
}

// ---------------- Kernel 4: Morse MLP (2x WMMA) + softplus -> [E,3] ---------------------
__global__ void k_morse(const int* __restrict__ eidx, const int* __restrict__ pidx,
                        const float* __restrict__ mw1, const float* __restrict__ mb1,
                        const float* __restrict__ mw2, const float* __restrict__ mb2,
                        const float* __restrict__ rDe, const float* __restrict__ rAl,
                        const float* __restrict__ rR0,
                        const float* __restrict__ hq, float* __restrict__ out, int E) {
    const int lane = threadIdx.x & 31;
    const int wave = blockIdx.x * (blockDim.x >> 5) + (threadIdx.x >> 5);
    const int base = wave * 32;
    if (base >= E) return;
    const int hi = lane >> 4;

    v16h A1 = wload_A16(mw1, lane, 16);
    v16h A2 = wload_A16(mw2, lane, 3);          // rows 3..15 zero

    int e  = base + lane;
    int sn = eidx[e];
    int tn = eidx[E + e];
    const float4* sp = (const float4*)(hq + (size_t)sn * D16);
    const float4* tp = (const float4*)(hq + (size_t)tn * D16);
    float pf[16];
#pragma unroll
    for (int q = 0; q < 4; ++q) {
        float4 a = sp[q], b = tp[q];
        pf[q * 4 + 0] = a.x + b.x; pf[q * 4 + 1] = a.y + b.y;
        pf[q * 4 + 2] = a.z + b.z; pf[q * 4 + 3] = a.w + b.w;
    }
    v16h B0 = pack_B16(pf);
    v16h B1 = shfl_B(B0, lane);

    float b1v[8];
#pragma unroll
    for (int j = 0; j < 8; ++j) b1v[j] = mb1[hi * 8 + j];

#pragma unroll
    for (int t = 0; t < 2; ++t) {
        v8f cz = {};
        v8f c1 = wmma16(A1, t ? B1 : B0, cz);
        float act[8];
#pragma unroll
        for (int j = 0; j < 8; ++j) act[j] = siluf(c1[j] + b1v[j]);
        v16h B2 = build_B2(act, lane);
        v8f c2 = wmma16(A2, B2, cz);            // delta^T: dims 0..2 at lanes<16, j=0..2

        if (lane < 16) {
            int e2 = base + t * 16 + lane;
            int p  = pidx[e2];
            float d0 = c2[0] + mb2[0];
            float d1 = c2[1] + mb2[1];
            float d2 = c2[2] + mb2[2];
            float De = softplusf(rDe[p] + d0);
            float al = softplusf(rAl[p] + d1);
            float r0 = rR0[p] + d2;
            float* o = out + (size_t)e2 * 3;
            o[0] = De; o[1] = al; o[2] = r0;
        }
    }
}

extern "C" void kernel_launch(void* const* d_in, const int* in_sizes, int n_in,
                              void* d_out, int out_size, void* d_ws, size_t ws_size,
                              hipStream_t stream) {
    (void)n_in; (void)out_size; (void)ws_size;
    const int*   elem  = (const int*)d_in[0];
    const int*   eidx  = (const int*)d_in[1];
    const float* dist  = (const float*)d_in[2];
    const int*   pidx  = (const int*)d_in[3];
    const float* embed = (const float*)d_in[4];
    const float* fw1   = (const float*)d_in[5];
    const float* fb1   = (const float*)d_in[6];
    const float* fw2   = (const float*)d_in[7];
    const float* fb2   = (const float*)d_in[8];
    const float* mw1   = (const float*)d_in[9];
    const float* mb1   = (const float*)d_in[10];
    const float* mw2   = (const float*)d_in[11];
    const float* mb2   = (const float*)d_in[12];
    const float* cbook = (const float*)d_in[13];
    const float* rDe   = (const float*)d_in[14];
    const float* rAl   = (const float*)d_in[15];
    const float* rR0   = (const float*)d_in[16];

    const int N = in_sizes[0];           // 100000
    const int E = in_sizes[2];           // 3200000 (divisible by 32)

    float* out  = (float*)d_out;
    float* h0   = (float*)d_ws;                  // [N,16]
    float* hacc = h0 + (size_t)N * D16;          // [N,16]
    float* hq   = hacc + (size_t)N * D16;        // [N,16]
    float* loss = out + (size_t)3 * E;           // scalar slot

    dim3 blk(256);
    int nodeBlocks = (N + 255) / 256;
    int edgeWaves  = (E + 31) / 32;
    int edgeBlocks = (edgeWaves + 7) / 8;        // 8 waves per block

    k_init  <<<nodeBlocks, blk, 0, stream>>>(elem, embed, h0, hacc, loss, N);
    k_filter<<<edgeBlocks, blk, 0, stream>>>(eidx, dist, fw1, fb1, fw2, fb2, h0, hacc, E);
    k_vq    <<<nodeBlocks, blk, 0, stream>>>(elem, cbook, hacc, hq, loss, N);
    k_morse <<<edgeBlocks, blk, 0, stream>>>(eidx, pidx, mw1, mb1, mw2, mb2,
                                             rDe, rAl, rR0, hq, out, E);
}
